// NaiveKmeans_25280177504397
// MI455X (gfx1250) — matrix operand order repaired and checked by reading, not compiled, not run on hardware
//
#include <hip/hip_runtime.h>
#include <hip/hip_bf16.h>

typedef __attribute__((ext_vector_type(16))) __bf16  v16bf;
typedef __attribute__((ext_vector_type(8)))  float   v8f;
typedef __attribute__((ext_vector_type(4)))  unsigned int u32x4;
typedef __attribute__((ext_vector_type(8)))  int     i32x8;
typedef __attribute__((ext_vector_type(4)))  int     i32x4;

#define N_PTS 131072
#define K_CTR 1024
#define DIM   128

// round-to-nearest-even f32 -> bf16 bits
static __device__ inline unsigned short f2bf(float f) {
    unsigned u = __float_as_uint(f);
    return (unsigned short)((u + 0x7FFFu + ((u >> 16) & 1u)) >> 16);
}

// ---------------- prep: centers f32 -> bf16, and ||c||^2 (fp32) ----------------
__global__ void kmeans_prep(const float* __restrict__ centers,
                            unsigned short* __restrict__ cb,
                            float* __restrict__ c2) {
    __shared__ float red[DIM];
    const int k = blockIdx.x;
    const int t = threadIdx.x;
    float v = centers[(size_t)k * DIM + t];
    cb[(size_t)k * DIM + t] = f2bf(v);
    red[t] = v * v;
    __syncthreads();
    for (int s = DIM / 2; s > 0; s >>= 1) {
        if (t < s) red[t] += red[t + s];
        __syncthreads();
    }
    if (t == 0) c2[k] = red[0];
}

// load one 32x16 B tile chunk set (4 q-chunks) for center tile t, ISA B layout
static __device__ inline void loadB(v16bf b[4], const unsigned short* __restrict__ cl,
                                    int t, int ln, int half) {
    const unsigned short* bp = cl + (size_t)(t * 16 + ln) * DIM + half * 8;
#pragma unroll
    for (int q = 0; q < 4; ++q) {
        union { v16bf v; uint4 u[2]; } bv;
        bv.u[0] = *(const uint4*)(bp + q * 32);
        bv.u[1] = *(const uint4*)(bp + q * 32 + 16);
        b[q] = bv.v;
    }
}

// ---------------- main: WMMA x@c^T + min/argmin ----------------
__launch_bounds__(256, 1)
__global__ void kmeans_assign(const float* __restrict__ x,
                              const unsigned short* __restrict__ cb,
                              const float* __restrict__ c2,
                              float* __restrict__ costs,
                              int* __restrict__ idxs) {
    extern __shared__ char smem[];
    unsigned short* cl = (unsigned short*)smem;                         // 256 KB bf16 centers
    float* x2s = (float*)(smem + (size_t)K_CTR * DIM * sizeof(unsigned short)); // 128 f32
    float* c2l = x2s + 128;                                             // 1024 f32 ||c||^2

    const int tid  = threadIdx.x;
    const int wave = tid >> 5;
    const int lane = tid & 31;
    const int half = lane >> 4;   // 0: lanes 0-15, 1: lanes 16-31
    const int ln   = lane & 15;

    // ---- stage all bf16 centers into LDS via Tensor Data Mover (one descriptor) ----
#if __has_builtin(__builtin_amdgcn_tensor_load_to_lds)
    if (wave == 0) {
        unsigned lds_base = __builtin_amdgcn_groupstaticsize();       // dynamic LDS starts here (0)
        unsigned long long ga = (unsigned long long)cb;
        u32x4 g0;
        g0[0] = 1u;                                                   // count=1, user mode
        g0[1] = lds_base;                                             // lds_addr (bytes)
        g0[2] = (unsigned)(ga & 0xFFFFFFFFu);                         // global_addr[31:0]
        g0[3] = (unsigned)((ga >> 32) & 0x01FFFFFFu) | (2u << 30);    // global_addr[56:32] | type=2
        i32x8 g1;
        g1[0] = (1 << 16);          // data_size = 1 -> 2 bytes
        g1[1] = (DIM << 16);        // tensor_dim0[15:0] = 128
        g1[2] = (K_CTR << 16);      // tensor_dim1[15:0] = 1024
        g1[3] = (DIM << 16);        // tile_dim0 = 128
        g1[4] = K_CTR;              // tile_dim1 = 1024 (tile_dim2 = 0)
        g1[5] = DIM;                // tensor_dim0_stride = 128 elements
        g1[6] = 0;
        g1[7] = 0;
        i32x4 z4 = {0, 0, 0, 0};
        i32x8 z8 = {0, 0, 0, 0, 0, 0, 0, 0};
        __builtin_amdgcn_tensor_load_to_lds(g0, g1, z4, z4, z8, 0);
        __builtin_amdgcn_s_wait_tensorcnt(0);
    }
#else
    for (int i = tid * 8; i < K_CTR * DIM; i += 256 * 8)
        *(uint4*)(cl + i) = *(const uint4*)(cb + i);
#endif

    // ---- stage ||c||^2 into LDS (4 KB): one float4 per thread ----
    ((float4*)c2l)[tid] = ((const float4*)c2)[tid];

    // ---- preload this wave's 16x128 A tile (bf16, ISA A-layout) + fp32 ||x||^2 ----
    const int row0  = blockIdx.x * 128 + wave * 16;
    const float* xr = x + (size_t)(row0 + ln) * DIM;

    v16bf a[4];
    float ss = 0.f;
#pragma unroll
    for (int q = 0; q < 4; ++q) {
        const int k1 = q * 32 + half * 8;        // K = k1..k1+7  -> VGPR 0-3 half
        const int k2 = q * 32 + 16 + half * 8;   // K = k2..k2+7  -> VGPR 4-7 half
        float4 f0 = *(const float4*)(xr + k1);
        float4 f1 = *(const float4*)(xr + k1 + 4);
        float4 f2 = *(const float4*)(xr + k2);
        float4 f3 = *(const float4*)(xr + k2 + 4);
        ss += f0.x*f0.x + f0.y*f0.y + f0.z*f0.z + f0.w*f0.w;
        ss += f1.x*f1.x + f1.y*f1.y + f1.z*f1.z + f1.w*f1.w;
        ss += f2.x*f2.x + f2.y*f2.y + f2.z*f2.z + f2.w*f2.w;
        ss += f3.x*f3.x + f3.y*f3.y + f3.z*f3.z + f3.w*f3.w;
        union { v16bf v; unsigned short s[16]; } av;
        av.s[0]  = f2bf(f0.x); av.s[1]  = f2bf(f0.y); av.s[2]  = f2bf(f0.z); av.s[3]  = f2bf(f0.w);
        av.s[4]  = f2bf(f1.x); av.s[5]  = f2bf(f1.y); av.s[6]  = f2bf(f1.z); av.s[7]  = f2bf(f1.w);
        av.s[8]  = f2bf(f2.x); av.s[9]  = f2bf(f2.y); av.s[10] = f2bf(f2.z); av.s[11] = f2bf(f2.w);
        av.s[12] = f2bf(f3.x); av.s[13] = f2bf(f3.y); av.s[14] = f2bf(f3.z); av.s[15] = f2bf(f3.w);
        a[q] = av.v;
    }
    // row ln's ||x||^2 = this lane's half + partner lane's half (lane ^ 16)
    float x2full = ss + __shfl_xor(ss, 16);
    if (half == 0) x2s[wave * 16 + ln] = x2full;

    __syncthreads();   // centers + c2 in LDS, x2s visible to all waves

    float x2v[8];
#pragma unroll
    for (int r = 0; r < 8; ++r) x2v[r] = x2s[wave * 16 + half * 8 + r];

    // ---- sweep all 64 center tiles, double-buffered B ----
    // Running value is m = ||c||^2 - 2*x.c ; per-row ||x||^2 and the >=0 clamp are
    // monotone/constant per row, so min/argmin are preserved; applied at the end.
    float runc[8];
    int   runi[8];
#pragma unroll
    for (int r = 0; r < 8; ++r) { runc[r] = 3.4e38f; runi[r] = 0; }

    v16bf b0[4], b1[4];
    loadB(b0, cl, 0, ln, half);

    for (int t = 0; t < K_CTR / 16; t += 2) {
        // ---- tile t (uses b0), prefetch tile t+1 into b1 ----
        loadB(b1, cl, t + 1, ln, half);
        {
            const int cidx = t * 16 + ln;
            const float cc = c2l[cidx];
            v8f acc = (v8f)0.0f;
#pragma unroll
            for (int q = 0; q < 4; ++q)
                acc = __builtin_amdgcn_wmma_f32_16x16x32_bf16(
                    false, a[q], false, b0[q], (short)0, acc, false, false);
#pragma unroll
            for (int r = 0; r < 8; ++r) {
                float m = fmaf(-2.0f, acc[r], cc);
                if (m < runc[r]) { runc[r] = m; runi[r] = cidx; }
            }
        }
        // ---- tile t+1 (uses b1), prefetch tile t+2 into b0 ----
        if (t + 2 < K_CTR / 16) loadB(b0, cl, t + 2, ln, half);
        {
            const int cidx = (t + 1) * 16 + ln;
            const float cc = c2l[cidx];
            v8f acc = (v8f)0.0f;
#pragma unroll
            for (int q = 0; q < 4; ++q)
                acc = __builtin_amdgcn_wmma_f32_16x16x32_bf16(
                    false, a[q], false, b1[q], (short)0, acc, false, false);
#pragma unroll
            for (int r = 0; r < 8; ++r) {
                float m = fmaf(-2.0f, acc[r], cc);
                if (m < runc[r]) { runc[r] = m; runi[r] = cidx; }
            }
        }
    }

    // ---- reduce min/argmin across the 16-lane N dimension (ties -> smallest index) ----
#pragma unroll
    for (int m = 8; m >= 1; m >>= 1) {
#pragma unroll
        for (int r = 0; r < 8; ++r) {
            float oc = __shfl_xor(runc[r], m);
            int   oi = __shfl_xor(runi[r], m);
            if (oc < runc[r] || (oc == runc[r] && oi < runi[r])) {
                runc[r] = oc; runi[r] = oi;
            }
        }
    }

    if (ln == 0) {
#pragma unroll
        for (int r = 0; r < 8; ++r) {
            int orow = row0 + half * 8 + r;
            costs[orow] = fmaxf(x2v[r] + runc[r], 0.0f);
            idxs[orow]  = runi[r];
        }
    }
}

extern "C" void kernel_launch(void* const* d_in, const int* in_sizes, int n_in,
                              void* d_out, int out_size, void* d_ws, size_t ws_size,
                              hipStream_t stream) {
    const float* x       = (const float*)d_in[0];
    const float* centers = (const float*)d_in[1];

    unsigned short* cb = (unsigned short*)d_ws;                          // 256 KB bf16 centers
    float* c2 = (float*)((char*)d_ws + (size_t)K_CTR * DIM * sizeof(unsigned short)); // 4 KB

    float* costs = (float*)d_out;
    int*   idxs  = (int*)d_out + N_PTS;

    kmeans_prep<<<K_CTR, DIM, 0, stream>>>(centers, cb, c2);

    size_t shmem = (size_t)K_CTR * DIM * sizeof(unsigned short)
                 + 128 * sizeof(float) + K_CTR * sizeof(float);
    (void)hipFuncSetAttribute(reinterpret_cast<const void*>(kmeans_assign),
                              hipFuncAttributeMaxDynamicSharedMemorySize, (int)shmem);
    kmeans_assign<<<N_PTS / 128, 256, shmem, stream>>>(x, cb, c2, costs, idxs);
}
